// MultiheadAttention_62139586838574
// MI455X (gfx1250) — matrix operand order repaired and checked by reading, not compiled, not run on hardware
//
#include <hip/hip_runtime.h>
#include <hip/hip_bf16.h>

#define B_    2
#define T_    2048
#define C_    2048
#define H_    16
#define KVH_  4
#define HD_   128
#define G_    (H_ / KVH_)
#define NQ_   (H_ * HD_)                 // 2048
#define NKV_  (KVH_ * HD_)               // 512
#define NQKV_ (NQ_ + 2 * NKV_)           // 3072
#define MTOK_ (B_ * T_)                  // 4096

typedef __bf16 bf16;
typedef __attribute__((ext_vector_type(16))) __bf16 v16bf;
typedef __attribute__((ext_vector_type(8)))  float  v8f;
typedef unsigned int u32;
typedef unsigned short u16;

// ---------------------------------------------------------------- WMMA core
static __device__ inline v8f wmma_bf16(v16bf a, v16bf b, v8f c) {
  // D(16x16,f32) = A(16x32,bf16) x B(32x16,bf16) + C
  return __builtin_amdgcn_wmma_f32_16x16x32_bf16(
      /*neg_a=*/false, a, /*neg_b=*/false, b,
      /*c_mod=*/(short)0, c, /*reuse_a=*/false, /*reuse_b=*/false);
}

// A fragment 16x32 (MxK); element (m,k) = p[m*ld + k].  Contiguous 16B runs ->
// lowers to 2x ds_load_b128 / global_load_b128.
static __device__ inline v16bf load_A_frag(const bf16* p, int ld) {
  const int lane = threadIdx.x & 31;
  const int m  = lane & 15;
  const int kb = (lane >> 4) << 4;   // 0 or 16
  const bf16* row = p + (size_t)m * ld;
  v16bf a;
  u32* u = reinterpret_cast<u32*>(&a);
#pragma unroll
  for (int j = 0; j < 4; ++j) {
    u[j]     = *reinterpret_cast<const u32*>(row + kb + 2 * j);        // K = kb+2j
    u[4 + j] = *reinterpret_cast<const u32*>(row + kb + 8 + 2 * j);    // K = kb+8+2j
  }
  return a;
}

// B fragment 32x16 (KxN) from k-contiguous storage: element (k,n) = p[n*ld + k].
// Per lane: one contiguous 32B run -> 2x ds_load_b128.
static __device__ inline v16bf load_B_kcontig(const bf16* p, int ld) {
  const int lane = threadIdx.x & 31;
  const int n  = lane & 15;
  const int kb = (lane >> 4) << 4;   // 0 or 16
  const bf16* col = p + (size_t)n * ld + kb;
  v16bf b;
  u32* u = reinterpret_cast<u32*>(&b);
#pragma unroll
  for (int j = 0; j < 8; ++j) u[j] = *reinterpret_cast<const u32*>(col + 2 * j);
  return b;
}

// ---------------------------------------------------------------- GEMM
// C[M,N](f32) = A[M,K](bf16,row-major) * Bw[K,N](bf16,row-major)
// block tile 128x128, 8 waves (4x2), 32x64 per wave, K-step 32.
// Register-staged double buffering: global loads for tile i+1 issued before the
// WMMA block, LDS stores after it -> loads stay outstanding across the math.
// B staged transposed ([n][k]) with dword-packed k-pairs for the frag loads.
__global__ __launch_bounds__(256) void gemm_bf16_f32(
    const bf16* __restrict__ A, const bf16* __restrict__ Bw,
    float* __restrict__ Cc, int M, int N, int K) {
  __shared__ bf16 As[2][128 * 32];
  __shared__ bf16 Bs[2][128 * 32];   // transposed: [n][k]
  const int m0 = blockIdx.x * 128;
  const int n0 = blockIdx.y * 128;
  const int t  = threadIdx.x;
  const int wave = t >> 5;
  const int wm = wave >> 1;          // 0..3 -> rows wm*32
  const int wn = wave & 1;           // 0..1 -> cols wn*64
  v8f acc[2][4] = {};

  const int arow = t >> 1, acol = (t & 1) * 16;   // A: 2 thr/row, 16 elems
  const int bkp  = t >> 4;                        // B: k-pair 0..15 (k = 2*bkp)
  const int bn0  = (t & 15) * 8;                  // B: 8 n-columns

  uint4 ra0, ra1, rb0, rb1;

  auto gload = [&](int k0) {
    const uint4* ga = reinterpret_cast<const uint4*>(A + (size_t)(m0 + arow) * K + k0 + acol);
    ra0 = ga[0];
    ra1 = ga[1];
    rb0 = *reinterpret_cast<const uint4*>(Bw + (size_t)(k0 + 2 * bkp) * N + n0 + bn0);
    rb1 = *reinterpret_cast<const uint4*>(Bw + (size_t)(k0 + 2 * bkp + 1) * N + n0 + bn0);
    __builtin_prefetch(A + (size_t)(m0 + arow) * K + k0 + 32 + acol, 0, 0);
    __builtin_prefetch(Bw + (size_t)(k0 + 32 + 2 * bkp) * N + n0 + bn0, 0, 0);
  };

  auto sstore = [&](int buf) {
    uint4* sa = reinterpret_cast<uint4*>(&As[buf][arow * 32 + acol]);
    sa[0] = ra0;
    sa[1] = ra1;
    const u16* lo = reinterpret_cast<const u16*>(&rb0);
    const u16* hi = reinterpret_cast<const u16*>(&rb1);
    u32* bs32 = reinterpret_cast<u32*>(&Bs[buf][0]);   // (n,k-pair) dwords
#pragma unroll
    for (int e = 0; e < 8; ++e)
      bs32[(bn0 + e) * 16 + bkp] = (u32)lo[e] | ((u32)hi[e] << 16);
  };

  gload(0);
  sstore(0);
  __syncthreads();

  const int niter = K >> 5;
  for (int it = 0; it < niter; ++it) {
    const bool has_next = (it + 1 < niter);
    if (has_next) gload((it + 1) << 5);          // issue global loads early
    const bf16* as = &As[it & 1][0];
    const bf16* bs = &Bs[it & 1][0];
    v16bf a0 = load_A_frag(as + (wm * 32 + 0) * 32, 32);
    v16bf a1 = load_A_frag(as + (wm * 32 + 16) * 32, 32);
#pragma unroll
    for (int j = 0; j < 4; ++j) {
      v16bf bj = load_B_kcontig(bs + (wn * 64 + j * 16) * 32, 32);
      acc[0][j] = wmma_bf16(a0, bj, acc[0][j]);
      acc[1][j] = wmma_bf16(a1, bj, acc[1][j]);
    }
    if (has_next) sstore((it + 1) & 1);          // LDS fill after the math
    __syncthreads();
  }

  const int lane = t & 31;
  const int nn = lane & 15;
  const int mb = (lane >> 4) * 8;    // C/D rows r / r+8 across lane halves
#pragma unroll
  for (int i = 0; i < 2; ++i)
#pragma unroll
    for (int j = 0; j < 4; ++j) {
      float* cp = Cc + (size_t)(m0 + wm * 32 + i * 16 + mb) * N + (n0 + wn * 64 + j * 16 + nn);
#pragma unroll
      for (int r = 0; r < 8; ++r) cp[(size_t)r * N] = acc[i][j][r];
    }
}

// ---------------------------------------------------------------- converts
__global__ void f32_to_bf16_kernel(const float* __restrict__ src, bf16* __restrict__ dst, size_t n) {
  size_t i = (size_t)blockIdx.x * blockDim.x + threadIdx.x;
  const size_t stride = (size_t)gridDim.x * blockDim.x;
  for (; i < n; i += stride) dst[i] = (bf16)src[i];
}

__global__ void build_wcat_kernel(const float* __restrict__ wq, const float* __restrict__ wk,
                                  const float* __restrict__ wv, bf16* __restrict__ Wcat) {
  const size_t n = (size_t)C_ * NQKV_;
  size_t i = (size_t)blockIdx.x * blockDim.x + threadIdx.x;
  const size_t stride = (size_t)gridDim.x * blockDim.x;
  for (; i < n; i += stride) {
    const int c  = (int)(i / NQKV_);
    const int nc = (int)(i % NQKV_);
    float v;
    if (nc < NQ_)             v = wq[(size_t)c * NQ_ + nc];
    else if (nc < NQ_ + NKV_) v = wk[(size_t)c * NKV_ + (nc - NQ_)];
    else                      v = wv[(size_t)c * NKV_ + (nc - NQ_ - NKV_)];
    Wcat[i] = (bf16)v;
  }
}

// ---------------------------------------------------------------- RoPE epilogue
// qkv[MTOK, 3072] f32 -> Qb[B*H,T,HD] / Kb,Vb[B*KVH,T,HD] bf16 (rope on q,k).
// Q is written pre-scaled by 1/sqrt(HD) so attention skips the multiply.
__global__ __launch_bounds__(256) void qkv_rope_kernel(
    const float* __restrict__ qkv, const float* __restrict__ bq,
    const float* __restrict__ bk, const float* __restrict__ bv,
    bf16* __restrict__ Qb, bf16* __restrict__ Kb, bf16* __restrict__ Vb) {
  const int tok = blockIdx.x;
  const int b = tok / T_, tp = tok % T_;
  const float* row = qkv + (size_t)tok * NQKV_;
  const float kln = 0.14391156831212787f;  // ln(10000)/64
  const float qscale = 0.08838834764831845f;  // 1/sqrt(128)
  for (int n = threadIdx.x; n < NQKV_; n += blockDim.x) {
    if (n < NQ_) {
      const int h = n >> 7, d = n & 127;
      const int fi = d & 63;
      const float ang = (float)tp * __expf(-(float)fi * kln);
      float s, c; __sincosf(ang, &s, &c);
      const int dp = (d < 64) ? d + 64 : d - 64;
      const float val = row[n] + bq[n];
      const float pv  = row[(h << 7) + dp] + bq[(h << 7) + dp];
      const float rot = (d < 64) ? -pv : pv;
      Qb[((size_t)(b * H_ + h) * T_ + tp) * HD_ + d] = (bf16)((val * c + rot * s) * qscale);
    } else if (n < NQ_ + NKV_) {
      const int nn = n - NQ_;
      const int h = nn >> 7, d = nn & 127;
      const int fi = d & 63;
      const float ang = (float)tp * __expf(-(float)fi * kln);
      float s, c; __sincosf(ang, &s, &c);
      const int dp = (d < 64) ? d + 64 : d - 64;
      const float val = row[n] + bk[nn];
      const float pv  = row[NQ_ + (h << 7) + dp] + bk[(h << 7) + dp];
      const float rot = (d < 64) ? -pv : pv;
      Kb[((size_t)(b * KVH_ + h) * T_ + tp) * HD_ + d] = (bf16)(val * c + rot * s);
    } else {
      const int nn = n - NQ_ - NKV_;
      const int h = nn >> 7, d = nn & 127;
      Vb[((size_t)(b * KVH_ + h) * T_ + tp) * HD_ + d] = (bf16)(row[n] + bv[nn]);
    }
  }
}

// ---------------------------------------------------------------- flash attention
// block = (b,h, 64 query rows), 4 waves x 16 rows; 32-key chunks.
// Register-staged double-buffered K/V LDS; V staged transposed ([d][key]) with
// dword-packed key pairs for contiguous B-fragment loads.
__global__ __launch_bounds__(128) void attn_kernel(
    const bf16* __restrict__ Qb,   // [B*H, T, HD], pre-scaled by 1/sqrt(HD)
    const bf16* __restrict__ Kb,   // [B*KVH, T, HD]
    const bf16* __restrict__ Vb,   // [B*KVH, T, HD]
    bf16* __restrict__ Ob) {       // [B*T, H*HD]
  __shared__ bf16 Ks[2][32 * HD_];     // [key][d]
  __shared__ bf16 Vt[2][HD_ * 32];     // [d][key]
  __shared__ bf16 Ps[4 * 16 * 32];

  const int bh = blockIdx.x;
  const int b  = bh / H_;
  const int h  = bh % H_;
  const int kv = h / G_;
  const int qbase = blockIdx.y * 64;

  const int t    = threadIdx.x;
  const int wvid = t >> 5;
  const int lane = t & 31;
  const int nn   = lane & 15;
  const int mb   = (lane >> 4) * 8;

  // Q rows for this wave, as 4 A-fragments covering HD=128
  const bf16* Qp = Qb + ((size_t)(b * H_ + h) * T_ + qbase + wvid * 16) * HD_;
  v16bf qf[4];
#pragma unroll
  for (int kk = 0; kk < 4; ++kk) qf[kk] = load_A_frag(Qp + kk * 32, HD_);

  v8f o[8] = {};
  float rowmax[8], rowsum[8];
#pragma unroll
  for (int r = 0; r < 8; ++r) { rowmax[r] = -3.0e38f; rowsum[r] = 0.f; }

  const size_t kvrow0 = (size_t)(b * KVH_ + kv) * T_;
  const int nchunks = (qbase + 64) >> 5;     // causal band

  // K staging map: row-major, 4x uint4 per thread
  const int krow = t >> 2;            // key row 0..31
  const int kseg = (t & 3) * 4;       // uint4 segment
  // V staging map: key pairs, 16 dims per thread
  const int vkp = t >> 3;             // key pair 0..15 (keys 2*vkp, 2*vkp+1)
  const int vd0 = (t & 7) * 16;       // dims vd0..vd0+15
  bf16* PsW = Ps + wvid * (16 * 32);

  uint4 rk[4], rvlo[2], rvhi[2];

  auto gload_kv = [&](int kc) {
    const uint4* gk = reinterpret_cast<const uint4*>(Kb + (kvrow0 + kc * 32 + krow) * HD_);
#pragma unroll
    for (int i = 0; i < 4; ++i) rk[i] = gk[kseg + i];
    const uint4* gv0 = reinterpret_cast<const uint4*>(Vb + (kvrow0 + kc * 32 + 2 * vkp) * HD_ + vd0);
    const uint4* gv1 = reinterpret_cast<const uint4*>(Vb + (kvrow0 + kc * 32 + 2 * vkp + 1) * HD_ + vd0);
    rvlo[0] = gv0[0]; rvlo[1] = gv0[1];
    rvhi[0] = gv1[0]; rvhi[1] = gv1[1];
  };

  auto sstore_kv = [&](int buf) {
    uint4* sk = reinterpret_cast<uint4*>(&Ks[buf][krow * HD_]);
#pragma unroll
    for (int i = 0; i < 4; ++i) sk[kseg + i] = rk[i];
    u32* vt32 = reinterpret_cast<u32*>(&Vt[buf][0]);   // (d, key-pair) dwords
#pragma unroll
    for (int j = 0; j < 2; ++j) {
      const u16* lo = reinterpret_cast<const u16*>(&rvlo[j]);
      const u16* hi = reinterpret_cast<const u16*>(&rvhi[j]);
#pragma unroll
      for (int e = 0; e < 8; ++e)
        vt32[(vd0 + j * 8 + e) * 16 + vkp] = (u32)lo[e] | ((u32)hi[e] << 16);
    }
  };

  gload_kv(0);
  sstore_kv(0);
  __syncthreads();

  for (int kc = 0; kc < nchunks; ++kc) {
    const bool has_next = (kc + 1 < nchunks);
    if (has_next) gload_kv(kc + 1);            // issue global loads early
    const bf16* ks = &Ks[kc & 1][0];
    const bf16* vt = &Vt[kc & 1][0];

    // S = Q * K^T : two 16-key column groups, K-dim 128 = 4 WMMA steps each
    v8f s0 = {}, s1 = {};
#pragma unroll
    for (int kk = 0; kk < 4; ++kk) {
      v16bf bk0 = load_B_kcontig(ks + kk * 32, HD_);              // keys 0..15
      v16bf bk1 = load_B_kcontig(ks + 16 * HD_ + kk * 32, HD_);   // keys 16..31
      s0 = wmma_bf16(qf[kk], bk0, s0);
      s1 = wmma_bf16(qf[kk], bk1, s1);
    }

    // online softmax (rows r / r+8 across lane halves; 16 lanes = 16 key cols)
#pragma unroll
    for (int r = 0; r < 8; ++r) {
      const int qrow = qbase + wvid * 16 + mb + r;
      const int key0 = kc * 32 + nn;
      float v0 = s0[r];
      float v1 = s1[r];
      if (key0 > qrow)      v0 = -3.0e38f;
      if (key0 + 16 > qrow) v1 = -3.0e38f;
      float mx = fmaxf(v0, v1);
#pragma unroll
      for (int off = 8; off >= 1; off >>= 1) mx = fmaxf(mx, __shfl_xor(mx, off, 16));
      const float mnew  = fmaxf(rowmax[r], mx);
      const float alpha = __expf(rowmax[r] - mnew);
      const float p0 = __expf(v0 - mnew);
      const float p1 = __expf(v1 - mnew);
      float ps = p0 + p1;
#pragma unroll
      for (int off = 8; off >= 1; off >>= 1) ps += __shfl_xor(ps, off, 16);
      rowsum[r] = rowsum[r] * alpha + ps;
      rowmax[r] = mnew;
#pragma unroll
      for (int d = 0; d < 8; ++d) o[d][r] *= alpha;
      PsW[(mb + r) * 32 + nn]      = (bf16)p0;   // stage P for A-layout reload
      PsW[(mb + r) * 32 + 16 + nn] = (bf16)p1;
    }

    // O += P(16x32) * V(32x128); V fragments are k-contiguous in Vt
    v16bf pf = load_A_frag(PsW, 32);
#pragma unroll
    for (int d = 0; d < 8; ++d) {
      v16bf vb = load_B_kcontig(vt + (d * 16) * 32, 32);
      o[d] = wmma_bf16(pf, vb, o[d]);
    }

    if (has_next) sstore_kv((kc + 1) & 1);     // LDS fill after the math
    __syncthreads();
  }

  // normalize and store token-major bf16 for the output-projection GEMM
#pragma unroll
  for (int r = 0; r < 8; ++r) {
    const float inv = 1.0f / rowsum[r];
    const int qrow = qbase + wvid * 16 + mb + r;
    bf16* op = Ob + (size_t)(b * T_ + qrow) * NQ_ + h * HD_;
#pragma unroll
    for (int d = 0; d < 8; ++d) op[d * 16 + nn] = (bf16)(o[d][r] * inv);
  }
}

// ---------------------------------------------------------------- launch
extern "C" void kernel_launch(void* const* d_in, const int* in_sizes, int n_in,
                              void* d_out, int out_size, void* d_ws, size_t ws_size,
                              hipStream_t stream) {
  (void)in_sizes; (void)n_in; (void)out_size; (void)ws_size;
  const float* x  = (const float*)d_in[0];
  const float* wq = (const float*)d_in[1];
  const float* bq = (const float*)d_in[2];
  const float* wk = (const float*)d_in[3];
  const float* bk = (const float*)d_in[4];
  const float* wv = (const float*)d_in[5];
  const float* bv = (const float*)d_in[6];
  const float* wo = (const float*)d_in[7];
  float* out = (float*)d_out;

  char* ws = (char*)d_ws;
  size_t off = 0;
  bf16*  xb   = (bf16*)(ws + off);  off += (size_t)MTOK_ * C_ * 2;           // 16 MB
  bf16*  Wcat = (bf16*)(ws + off);  off += (size_t)C_ * NQKV_ * 2;           // 12 MB
  bf16*  Wob  = (bf16*)(ws + off);  off += (size_t)NQ_ * C_ * 2;             //  8 MB
  float* qkv  = (float*)(ws + off); off += (size_t)MTOK_ * NQKV_ * 4;        // 48 MB
  bf16*  Qb   = (bf16*)(ws + off);  off += (size_t)B_ * H_ * T_ * HD_ * 2;   // 16 MB
  bf16*  Kb   = (bf16*)(ws + off);  off += (size_t)B_ * KVH_ * T_ * HD_ * 2; //  4 MB
  bf16*  Vb   = (bf16*)(ws + off);  off += (size_t)B_ * KVH_ * T_ * HD_ * 2; //  4 MB
  bf16*  Ob   = (bf16*)(ws + off);  off += (size_t)MTOK_ * NQ_ * 2;          // 16 MB

  f32_to_bf16_kernel<<<4096, 256, 0, stream>>>(x, xb, (size_t)MTOK_ * C_);
  build_wcat_kernel<<<4096, 256, 0, stream>>>(wq, wk, wv, Wcat);
  f32_to_bf16_kernel<<<4096, 256, 0, stream>>>(wo, Wob, (size_t)NQ_ * C_);

  gemm_bf16_f32<<<dim3(MTOK_ / 128, NQKV_ / 128), 256, 0, stream>>>(
      xb, Wcat, qkv, MTOK_, NQKV_, C_);
  qkv_rope_kernel<<<MTOK_, 256, 0, stream>>>(qkv, bq, bk, bv, Qb, Kb, Vb);
  attn_kernel<<<dim3(B_ * H_, T_ / 64), 128, 0, stream>>>(Qb, Kb, Vb, Ob);
  gemm_bf16_f32<<<dim3(MTOK_ / 128, C_ / 128), 256, 0, stream>>>(
      Ob, Wob, out, MTOK_, C_, NQ_);
}
